// GloRe_Unit_23313082482943
// MI455X (gfx1250) — matrix operand order repaired and checked by reading, not compiled, not run on hardware
//
#include <hip/hip_runtime.h>
#include <hip/hip_bf16.h>

// ---------------- problem constants ----------------
constexpr int kB    = 16;
constexpr int kC    = 256;
constexpr int kHW   = 96 * 96;   // 9216
constexpr int kS    = 128;
constexpr int kN    = 64;
constexpr int kKCH  = 32;        // K-split chunks for the graph-projection GEMM
constexpr float kEPS = 1e-5f;

typedef __bf16 bf16_t;
typedef bf16_t v16bf __attribute__((ext_vector_type(16)));
typedef float  v8f   __attribute__((ext_vector_type(8)));

// LDS B-tile row pitch for K1: kC elements + 8 pad -> 528B row, lane stride
// 132 dwords => 4-bank step => each 16-lane half-group's b128 reads cover all
// 64 banks exactly once (even 2-way for the full wave, the wave32 optimum).
constexpr int kBpitch = kC + 8;

// ---------------- workspace layout (bytes) ----------------
constexpr size_t WS_WALL   = 0;          // bf16 [256][256] packed (w_s;w_p;w_r)
constexpr size_t WS_ALPHA  = 131072;     // f32 [256] folded proj conv-bias+BN
constexpr size_t WS_BETA   = 132096;
constexpr size_t WS_ALPHAO = 133120;     // f32 [256] folded output BN
constexpr size_t WS_BETAO  = 134144;
constexpr size_t WS_XS     = 262144;                               // bf16 [B][128][9216]
constexpr size_t WS_XP     = WS_XS  + (size_t)kB * kS * kHW * 2;   // bf16 [B][64][9216]
constexpr size_t WS_XRT    = WS_XP  + (size_t)kB * kN * kHW * 2;   // bf16 [B][9216][64]
constexpr size_t WS_GPART  = WS_XRT + (size_t)kB * kHW * kN * 2;   // f32 [KCH][B][128][64]
constexpr size_t WS_GF     = WS_GPART + (size_t)kKCH * kB * kS * kN * 4; // bf16 [B][256][64]

// ---------------- fragment helpers ----------------
__device__ inline v8f zero8() {
    v8f z = {0.f, 0.f, 0.f, 0.f, 0.f, 0.f, 0.f, 0.f};
    return z;
}

__device__ inline v8f wmma_bf16(v16bf a, v16bf b, v8f c) {
    return __builtin_amdgcn_wmma_f32_16x16x32_bf16(false, a, false, b, (short)0, c,
                                                   false, false);
}

// A (16x32, bf16, row-major source): lane<16 -> row=lane, K {0..15};
// lane>=16 -> row=lane-16, K {16..31}
__device__ inline v16bf load_a_frag(const bf16_t* __restrict__ base, int ld,
                                    int mtile_row, int k0, int lane) {
    int row = lane & 15;
    int klo = (lane < 16) ? 0 : 16;
    const bf16_t* p = base + (size_t)(mtile_row + row) * ld + k0 + klo;
    v16bf a;
#pragma unroll
    for (int i = 0; i < 16; ++i) a[i] = p[i];
    return a;
}

// B (32x16, bf16): lane holds column n = n0 + (lane&15); K contiguous per lane,
// half-groups cover K {0..15} / {16..31}. Source layout: base[n*ld + k].
__device__ inline v16bf load_b_frag_kfast(const bf16_t* base, int ld,
                                          int n0, int k0, int lane) {
    int n    = n0 + (lane & 15);
    int koff = (lane >> 4) * 16;
    const bf16_t* p = base + (size_t)n * ld + k0 + koff;
    v16bf b;
#pragma unroll
    for (int i = 0; i < 16; ++i) b[i] = p[i];
    return b;
}

// ---------------- K0: pack weights, fold BN ----------------
__global__ void k0_prep(const float* __restrict__ w_s, const float* __restrict__ b_s,
                        const float* __restrict__ sg, const float* __restrict__ sb,
                        const float* __restrict__ sm, const float* __restrict__ sv,
                        const float* __restrict__ w_p, const float* __restrict__ b_p,
                        const float* __restrict__ pg, const float* __restrict__ pb,
                        const float* __restrict__ pm, const float* __restrict__ pv,
                        const float* __restrict__ w_r, const float* __restrict__ b_r,
                        const float* __restrict__ rg, const float* __restrict__ rb,
                        const float* __restrict__ rm, const float* __restrict__ rv,
                        const float* __restrict__ og, const float* __restrict__ ob,
                        const float* __restrict__ om, const float* __restrict__ ov,
                        bf16_t* __restrict__ wall, float* __restrict__ alpha,
                        float* __restrict__ beta, float* __restrict__ alphaO,
                        float* __restrict__ betaO) {
    int idx = blockIdx.x * blockDim.x + threadIdx.x;
    if (idx < kC * kC) {
        int o = idx >> 8, c = idx & 255;
        float w = (o < kS) ? w_s[(size_t)o * kC + c]
                 : (o < kS + kN) ? w_p[(size_t)(o - kS) * kC + c]
                 : w_r[(size_t)(o - kS - kN) * kC + c];
        wall[idx] = (bf16_t)w;
    }
    if (idx < kC) {
        int o = idx;
        float g, v, m, bb, bc;
        if (o < kS)            { g = sg[o]; v = sv[o]; m = sm[o]; bb = sb[o]; bc = b_s[o]; }
        else if (o < kS + kN)  { int i = o - kS;      g = pg[i]; v = pv[i]; m = pm[i]; bb = pb[i]; bc = b_p[i]; }
        else                   { int i = o - kS - kN; g = rg[i]; v = rv[i]; m = rm[i]; bb = rb[i]; bc = b_r[i]; }
        float sc = g * rsqrtf(v + kEPS);
        alpha[o] = sc;
        beta[o]  = (bc - m) * sc + bb;
        float so = og[o] * rsqrtf(ov[o] + kEPS);
        alphaO[o] = so;
        betaO[o]  = ob[o] - om[o] * so;
    }
}

// ---------------- K1: fused 3-way projection GEMM + BN + ReLU ----------------
// Per block: [256 out-channels] x [64 hw cols], K=256.
// x-tile staged ONCE in LDS as bf16 [col][k] (K-fast); 8 waves, each 2M x 4N tiles.
__global__ void __launch_bounds__(256, 1)
k1_proj(const float* __restrict__ x, const bf16_t* __restrict__ wall,
        const float* __restrict__ alpha, const float* __restrict__ beta,
        bf16_t* __restrict__ xs, bf16_t* __restrict__ xp,
        bf16_t* __restrict__ xrt) {
    __shared__ bf16_t Bs[64 * kBpitch];   // 64 cols x 264 bf16 = 33 KB

    const int b    = blockIdx.y;
    const int hw0  = blockIdx.x * 64;
    const int tid  = threadIdx.x;
    const int wv   = tid >> 5;
    const int lane = tid & 31;

    // --- cooperative stage of x[b][:, hw0:hw0+64] -> LDS bf16, [col][k] layout.
    // 256 threads = 4 k-groups x 64 cols; global reads are 64-lane-contiguous f32.
    {
        int col   = tid & 63;
        int kbase = (tid >> 6) * 64;
        const float* xcol = x + (size_t)b * kC * kHW + hw0 + col;
        bf16_t* dst = Bs + (size_t)col * kBpitch + kbase;
#pragma unroll 4
        for (int i = 0; i < 64; i += 2) {
            float f0 = xcol[(size_t)(kbase + i) * kHW];
            float f1 = xcol[(size_t)(kbase + i + 1) * kHW];
            dst[i]     = (bf16_t)f0;
            dst[i + 1] = (bf16_t)f1;
        }
    }
    __syncthreads();

    v8f acc[2][4];
    for (int i = 0; i < 2; ++i)
        for (int j = 0; j < 4; ++j) acc[i][j] = zero8();

    for (int k0 = 0; k0 < kC; k0 += 32) {
        v16bf a0 = load_a_frag(wall, kC, (2 * wv) * 16, k0, lane);
        v16bf a1 = load_a_frag(wall, kC, (2 * wv + 1) * 16, k0, lane);
        v16bf bf[4];
#pragma unroll
        for (int j = 0; j < 4; ++j)
            bf[j] = load_b_frag_kfast(Bs, kBpitch, j * 16, k0, lane);
#pragma unroll
        for (int j = 0; j < 4; ++j) {
            acc[0][j] = wmma_bf16(a0, bf[j], acc[0][j]);
            acc[1][j] = wmma_bf16(a1, bf[j], acc[1][j]);
        }
    }

    // Branch-free epilogue: destination region is uniform per (wave,i) tile
    // because the xs/xp/xr split points (128,192) are 16-row aligned.
    const int rowhalf = (lane >> 4) * 8;
    const int ncol    = lane & 15;
    bf16_t* xs_b  = xs  + (size_t)b * kS * kHW;
    bf16_t* xp_b  = xp  + (size_t)b * kN * kHW;
    bf16_t* xrt_b = xrt + (size_t)b * kHW * kN;
#pragma unroll
    for (int i = 0; i < 2; ++i) {
        const int tile  = 2 * wv + i;          // 0..15
        const bool to_s = tile < 8;
        const bool to_p = (!to_s) && (tile < 12);
        bf16_t* base = to_s ? xs_b : (to_p ? xp_b : xrt_b);
        const int moff = to_s ? 0 : (to_p ? kS : (kS + kN));
        const size_t strideM = (tile < 12) ? (size_t)kHW : (size_t)1;
        const size_t strideC = (tile < 12) ? (size_t)1 : (size_t)kN;
        bf16_t* rowp = base + (size_t)(tile * 16 + rowhalf - moff) * strideM
                            + (size_t)(hw0 + ncol) * strideC;
#pragma unroll
        for (int j = 0; j < 4; ++j) {
#pragma unroll
            for (int r = 0; r < 8; ++r) {
                int m = tile * 16 + rowhalf + r;
                float v = acc[i][j][r] * alpha[m] + beta[m];
                v = v > 0.f ? v : 0.f;
                rowp[(size_t)r * strideM + (size_t)(j * 16) * strideC] = (bf16_t)v;
            }
        }
    }
}

// ---------------- K2: g_part = xs . xp^T  (split-K, deterministic) ----------------
// Per block: full [128 x 64] tile over a 288-wide K chunk. 8 waves x (2M x 2N) tiles.
__global__ void k2_graph(const bf16_t* __restrict__ xs, const bf16_t* __restrict__ xp,
                         float* __restrict__ gpart) {
    const int b  = blockIdx.y;
    const int kc = blockIdx.x;
    const int wv = threadIdx.x >> 5;
    const int lane = threadIdx.x & 31;
    const bf16_t* xsb = xs + (size_t)b * kS * kHW;
    const bf16_t* xpb = xp + (size_t)b * kN * kHW;

    const int mt = 2 * (wv & 3);   // 8 M-tiles total
    const int nt = 2 * (wv >> 2);  // 4 N-tiles total

    v8f acc[2][2];
    for (int i = 0; i < 2; ++i)
        for (int j = 0; j < 2; ++j) acc[i][j] = zero8();

    const int kbeg = kc * (kHW / kKCH);
    const int kend = kbeg + kHW / kKCH;
    for (int k0 = kbeg; k0 < kend; k0 += 32) {
        v16bf a0 = load_a_frag(xsb, kHW, mt * 16, k0, lane);
        v16bf a1 = load_a_frag(xsb, kHW, (mt + 1) * 16, k0, lane);
        v16bf b0 = load_b_frag_kfast(xpb, kHW, nt * 16, k0, lane);
        v16bf b1 = load_b_frag_kfast(xpb, kHW, (nt + 1) * 16, k0, lane);
        acc[0][0] = wmma_bf16(a0, b0, acc[0][0]);
        acc[0][1] = wmma_bf16(a0, b1, acc[0][1]);
        acc[1][0] = wmma_bf16(a1, b0, acc[1][0]);
        acc[1][1] = wmma_bf16(a1, b1, acc[1][1]);
    }

    float* gp = gpart + ((size_t)kc * kB + b) * (kS * kN);
    const int rowhalf = (lane >> 4) * 8;
    const int ncol    = lane & 15;
    for (int i = 0; i < 2; ++i)
        for (int j = 0; j < 2; ++j) {
            int n = (nt + j) * 16 + ncol;
#pragma unroll
            for (int r = 0; r < 8; ++r) {
                int m = (mt + i) * 16 + rowhalf + r;
                gp[m * kN + n] = acc[i][j][r];
            }
        }
}

// ---------------- K3: reduce split-K, 2x GCN, fold fc2 -> Gf ----------------
__global__ void k3_gcn(const float* __restrict__ gpart,
                       const float* __restrict__ g1w1, const float* __restrict__ g1b1,
                       const float* __restrict__ g1w2,
                       const float* __restrict__ g2w1, const float* __restrict__ g2b1,
                       const float* __restrict__ g2w2,
                       const float* __restrict__ fc2w, bf16_t* __restrict__ gf) {
    __shared__ float gS[kS * kN];
    __shared__ float hS[kS * kN];
    const int b = blockIdx.x, tid = threadIdx.x;

    for (int e = tid; e < kS * kN; e += 256) {
        float s = 0.f;
        for (int kc = 0; kc < kKCH; ++kc)
            s += gpart[((size_t)kc * kB + b) * (kS * kN) + e];
        gS[e] = s * (1.0f / kHW);
    }
    __syncthreads();

    const float* W1[2] = {g1w1, g2w1};
    const float* B1[2] = {g1b1, g2b1};
    const float* W2[2] = {g1w2, g2w2};
    for (int layer = 0; layer < 2; ++layer) {
        const float* w1 = W1[layer];
        const float* b1 = B1[layer];
        const float* w2 = W2[layer];
        // node mixing + residual + LeakyReLU(0.2): h[s,m] = sum_n g[s,n]*w1[m,n] + b1[m] + g[s,m]
        for (int e = tid; e < kS * kN; e += 256) {
            int s = e >> 6, m = e & 63;
            float a = b1[m] + gS[e];
            for (int n = 0; n < kN; ++n) a += gS[(s << 6) + n] * w1[(m << 6) + n];
            hS[e] = a > 0.f ? a : 0.2f * a;
        }
        __syncthreads();
        // state mixing: g'[t,n] = sum_s w2[t,s]*h[s,n]
        for (int e = tid; e < kS * kN; e += 256) {
            int t = e >> 6, n = e & 63;
            float a = 0.f;
            for (int s = 0; s < kS; ++s) a += w2[t * kS + s] * hS[(s << 6) + n];
            gS[e] = a;
        }
        __syncthreads();
    }
    // Gf[c,n] = sum_s fc2_w[c,s] * g[s,n]
    for (int e = tid; e < kC * kN; e += 256) {
        int c = e >> 6, n = e & 63;
        float a = 0.f;
        for (int s = 0; s < kS; ++s) a += fc2w[c * kS + s] * gS[(s << 6) + n];
        gf[(size_t)b * kC * kN + e] = (bf16_t)a;
    }
}

// ---------------- K4: out = x + BN(Gf . xr) ----------------
// Per block: [256 channels] x [32 hw cols], K=64 (2 wmma steps).
__global__ void k4_out(const float* __restrict__ x, const bf16_t* __restrict__ gf,
                       const bf16_t* __restrict__ xrt,
                       const float* __restrict__ alphaO, const float* __restrict__ betaO,
                       float* __restrict__ out) {
    const int b   = blockIdx.y;
    const int hw0 = blockIdx.x * 32;
    const int wv  = threadIdx.x >> 5;
    const int lane = threadIdx.x & 31;
    const bf16_t* gfb = gf + (size_t)b * kC * kN;
    const bf16_t* xrb = xrt + (size_t)b * kHW * kN;

    v8f acc[2][2];
    for (int i = 0; i < 2; ++i)
        for (int j = 0; j < 2; ++j) acc[i][j] = zero8();

#pragma unroll
    for (int k0 = 0; k0 < kN; k0 += 32) {
        v16bf a0 = load_a_frag(gfb, kN, (2 * wv) * 16, k0, lane);
        v16bf a1 = load_a_frag(gfb, kN, (2 * wv + 1) * 16, k0, lane);
        v16bf b0 = load_b_frag_kfast(xrb, kN, hw0, k0, lane);
        v16bf b1 = load_b_frag_kfast(xrb, kN, hw0 + 16, k0, lane);
        acc[0][0] = wmma_bf16(a0, b0, acc[0][0]);
        acc[0][1] = wmma_bf16(a0, b1, acc[0][1]);
        acc[1][0] = wmma_bf16(a1, b0, acc[1][0]);
        acc[1][1] = wmma_bf16(a1, b1, acc[1][1]);
    }

    const int rowhalf = (lane >> 4) * 8;
    const int ncol    = lane & 15;
    for (int i = 0; i < 2; ++i)
        for (int j = 0; j < 2; ++j) {
            int col = hw0 + j * 16 + ncol;
#pragma unroll
            for (int r = 0; r < 8; ++r) {
                int c = (2 * wv + i) * 16 + rowhalf + r;
                size_t idx = ((size_t)b * kC + c) * (size_t)kHW + col;
                out[idx] = x[idx] + acc[i][j][r] * alphaO[c] + betaO[c];
            }
        }
}

// ---------------- host ----------------
extern "C" void kernel_launch(void* const* d_in, const int* in_sizes, int n_in,
                              void* d_out, int out_size, void* d_ws, size_t ws_size,
                              hipStream_t stream) {
    (void)in_sizes; (void)n_in; (void)out_size; (void)ws_size;
    const float* x    = (const float*)d_in[0];
    const float* w_s  = (const float*)d_in[1];
    const float* b_s  = (const float*)d_in[2];
    const float* sg   = (const float*)d_in[3];
    const float* sb   = (const float*)d_in[4];
    const float* sm   = (const float*)d_in[5];
    const float* sv   = (const float*)d_in[6];
    const float* w_p  = (const float*)d_in[7];
    const float* b_p  = (const float*)d_in[8];
    const float* pg   = (const float*)d_in[9];
    const float* pb   = (const float*)d_in[10];
    const float* pm   = (const float*)d_in[11];
    const float* pv   = (const float*)d_in[12];
    const float* w_r  = (const float*)d_in[13];
    const float* b_r  = (const float*)d_in[14];
    const float* rg   = (const float*)d_in[15];
    const float* rb   = (const float*)d_in[16];
    const float* rm   = (const float*)d_in[17];
    const float* rv   = (const float*)d_in[18];
    const float* g1w1 = (const float*)d_in[19];
    const float* g1b1 = (const float*)d_in[20];
    const float* g1w2 = (const float*)d_in[21];
    const float* g2w1 = (const float*)d_in[22];
    const float* g2b1 = (const float*)d_in[23];
    const float* g2w2 = (const float*)d_in[24];
    const float* fc2w = (const float*)d_in[25];
    const float* og   = (const float*)d_in[26];
    const float* ob   = (const float*)d_in[27];
    const float* om   = (const float*)d_in[28];
    const float* ov   = (const float*)d_in[29];

    char* ws = (char*)d_ws;
    bf16_t* wall   = (bf16_t*)(ws + WS_WALL);
    float*  alpha  = (float*)(ws + WS_ALPHA);
    float*  beta   = (float*)(ws + WS_BETA);
    float*  alphaO = (float*)(ws + WS_ALPHAO);
    float*  betaO  = (float*)(ws + WS_BETAO);
    bf16_t* xs     = (bf16_t*)(ws + WS_XS);
    bf16_t* xp     = (bf16_t*)(ws + WS_XP);
    bf16_t* xrt    = (bf16_t*)(ws + WS_XRT);
    float*  gpart  = (float*)(ws + WS_GPART);
    bf16_t* gf     = (bf16_t*)(ws + WS_GF);

    k0_prep<<<256, 256, 0, stream>>>(w_s, b_s, sg, sb, sm, sv,
                                     w_p, b_p, pg, pb, pm, pv,
                                     w_r, b_r, rg, rb, rm, rv,
                                     og, ob, om, ov,
                                     wall, alpha, beta, alphaO, betaO);
    k1_proj<<<dim3(kHW / 64, kB), 256, 0, stream>>>(x, wall, alpha, beta, xs, xp, xrt);
    k2_graph<<<dim3(kKCH, kB), 256, 0, stream>>>(xs, xp, gpart);
    k3_gcn<<<kB, 256, 0, stream>>>(gpart, g1w1, g1b1, g1w2, g2w1, g2b1, g2w2, fc2w, gf);
    k4_out<<<dim3(kHW / 32, kB), 256, 0, stream>>>(x, gf, xrt, alphaO, betaO, (float*)d_out);
}
